// ReconGNN_7894149890553
// MI455X (gfx1250) — compile-verified, matrix operand order checked by reading, not run on hardware
//
#include <hip/hip_runtime.h>
#include <hip/hip_bf16.h>

// ---------------------------------------------------------------------------
// ReconGNN forward for MI455X (gfx1250, wave32).
//  - All dense GEMMs: v_wmma_f32_16x16x32_bf16 (bf16 in, f32 accumulate).
//  - Activations live in bf16 (L2-resident, halves gather traffic);
//    scatter-add aggregation in fp32 via global_atomic_add_f32.
//  - LDS activation slabs staged with global_load_async_to_lds_b128 (ASYNCcnt).
// ---------------------------------------------------------------------------

#define NN 100000           // nodes  (divisible by 32)
#define NE 800000           // edges  (divisible by 64)
#define H  128
#define LDSTR 136           // LDS row stride in ushorts (16B-aligned rows, conflict padding)

typedef __attribute__((ext_vector_type(16))) __bf16 v16bf;
typedef __attribute__((ext_vector_type(8)))  float  v8f;

union Frag { v16bf v; uint4 q[2]; };
union Acc  { v8f   v; float f[8]; };

__device__ __forceinline__ unsigned short f2bf(float f) {
  union { float f; unsigned u; } x; x.f = f;
  unsigned r = x.u + 0x7FFFu + ((x.u >> 16) & 1u);   // round-to-nearest-even
  return (unsigned short)(r >> 16);
}
__device__ __forceinline__ float bf2f_lo(unsigned u) {
  union { unsigned u; float f; } v; v.u = u << 16; return v.f;
}
__device__ __forceinline__ float bf2f_hi(unsigned u) {
  union { unsigned u; float f; } v; v.u = u & 0xFFFF0000u; return v.f;
}

// --- gfx1250 async copy: 16 bytes global -> LDS, tracked by ASYNCcnt ---------
// LDS operand is the wave-relative LDS address = low 32 bits of the flat ptr.
__device__ __forceinline__ void async_b128(const void* gptr, void* lptr) {
  unsigned lds = (unsigned)(unsigned long long)lptr;
  asm volatile("global_load_async_to_lds_b128 %0, %1, off"
               :: "v"(lds), "v"(gptr) : "memory");
}
__device__ __forceinline__ void wait_async0() {
#if __has_builtin(__builtin_amdgcn_s_wait_asynccnt)
  __builtin_amdgcn_s_wait_asynccnt(0);
#else
  asm volatile("s_wait_asynccnt 0x0" ::: "memory");
#endif
}

// Build the 16x32 bf16 A-fragment for one lane from an LDS tile.
// ISA 7.12.2 (16-bit A, wave32): m = lane&15 ; k = (h&7) + 16*(h>>3) + 8*(lane>>4)
// => two contiguous 8-element (16B) runs per lane.
__device__ __forceinline__ v16bf load_a_frag(const unsigned short* lds, int lane, int kbase) {
  Frag a;
  int off = (lane & 15) * LDSTR + kbase + ((lane >> 4) << 3);
  a.q[0] = *(const uint4*)(lds + off);
  a.q[1] = *(const uint4*)(lds + off + 16);
  return a.v;
}

// ---------------------------------------------------------------------------
// Weight packing: row-major fp32 W[K][128] -> bf16 WMMA B-operand tiles.
// tile t = kt*8 + nt; dst[t*512 + lane*16 + h] = W[k][n],
// k = kt*32 + (h&7) + 16*(h>>3) + 8*(lane>>4), n = nt*16 + (lane&15).
// ---------------------------------------------------------------------------
__global__ void pack_kernel(const float* __restrict__ W, unsigned short* __restrict__ dst, int Ktiles) {
  int idx = blockIdx.x * 256 + threadIdx.x;
  if (idx >= Ktiles * 8 * 512) return;
  int tile = idx >> 9, within = idx & 511;
  int lane = within >> 4, h = within & 15;
  int kt = tile >> 3, nt = tile & 7;
  int k = kt * 32 + (h & 7) + ((h >> 3) << 4) + (((lane >> 4) & 1) << 3);
  int n = nt * 16 + (lane & 15);
  dst[idx] = f2bf(W[k * H + n]);
}

// ---------------------------------------------------------------------------
// Node encoder -> bf16 activations. Each thread produces a packed pair of cols.
// ---------------------------------------------------------------------------
__global__ void encode_kernel(const float* __restrict__ node_static,
                              const float* __restrict__ p_obs,
                              const int*   __restrict__ p_mask,
                              const float* __restrict__ W_enc,
                              const float* __restrict__ b_enc,
                              unsigned* __restrict__ xbf_pairs) {   // [NN][64] packed bf16x2
  int idx = blockIdx.x * 256 + threadIdx.x;
  if (idx >= NN * 64) return;
  int n = idx >> 6, cp = idx & 63, c = cp << 1;
  const float* ns = node_static + n * 6;
  float f6 = p_obs[n];
  float f7 = p_mask[n] ? 1.f : 0.f;
  float v0 = b_enc[c],     v1 = b_enc[c + 1];
#pragma unroll
  for (int k = 0; k < 6; ++k) {
    v0 += ns[k] * W_enc[k * H + c];
    v1 += ns[k] * W_enc[k * H + c + 1];
  }
  v0 += f6 * W_enc[6 * H + c] + f7 * W_enc[7 * H + c];
  v1 += f6 * W_enc[6 * H + c + 1] + f7 * W_enc[7 * H + c + 1];
  xbf_pairs[idx] = (unsigned)f2bf(v0) | ((unsigned)f2bf(v1) << 16);
}

__global__ void zero_kernel(float* __restrict__ p, int n4) {
  int i = blockIdx.x * 256 + threadIdx.x;
  if (i < n4) ((float4*)p)[i] = make_float4(0.f, 0.f, 0.f, 0.f);
}

__global__ void degree_kernel(const int* __restrict__ src, const int* __restrict__ dst,
                              float* __restrict__ deg) {
  int e = blockIdx.x * 256 + threadIdx.x;
  if (e < NE) {
    unsafeAtomicAdd(&deg[src[e]], 1.f);
    unsafeAtomicAdd(&deg[dst[e]], 1.f);
  }
}

__global__ void deginv_kernel(const float* __restrict__ deg, float* __restrict__ deg_inv) {
  int n = blockIdx.x * 256 + threadIdx.x;
  if (n < NN) deg_inv[n] = 1.f / fmaxf(deg[n], 1.f);
}

// Bidirectional scatter-add: agg[dst]+=x[src], agg[src]+=x[dst].
// 16 threads/edge; bf16 row gather (16B) -> 8 fp32 global atomics each way.
__global__ void aggregate_kernel(const unsigned short* __restrict__ xbf,
                                 const int* __restrict__ src, const int* __restrict__ dst,
                                 float* __restrict__ agg) {
  int idx = blockIdx.x * 256 + threadIdx.x;   // NE*16 = 12.8M
  int e = idx >> 4;
  if (e >= NE) return;
  int c = (idx & 15) << 3;
  int s = src[e], d = dst[e];
  uint4 us = *(const uint4*)(xbf + (size_t)s * H + c);
  uint4 ud = *(const uint4*)(xbf + (size_t)d * H + c);
  float* as = agg + (size_t)s * H + c;
  float* ad = agg + (size_t)d * H + c;
  unsafeAtomicAdd(ad + 0, bf2f_lo(us.x)); unsafeAtomicAdd(ad + 1, bf2f_hi(us.x));
  unsafeAtomicAdd(ad + 2, bf2f_lo(us.y)); unsafeAtomicAdd(ad + 3, bf2f_hi(us.y));
  unsafeAtomicAdd(ad + 4, bf2f_lo(us.z)); unsafeAtomicAdd(ad + 5, bf2f_hi(us.z));
  unsafeAtomicAdd(ad + 6, bf2f_lo(us.w)); unsafeAtomicAdd(ad + 7, bf2f_hi(us.w));
  unsafeAtomicAdd(as + 0, bf2f_lo(ud.x)); unsafeAtomicAdd(as + 1, bf2f_hi(ud.x));
  unsafeAtomicAdd(as + 2, bf2f_lo(ud.y)); unsafeAtomicAdd(as + 3, bf2f_hi(ud.y));
  unsafeAtomicAdd(as + 4, bf2f_lo(ud.z)); unsafeAtomicAdd(as + 5, bf2f_hi(ud.z));
  unsafeAtomicAdd(as + 6, bf2f_lo(ud.w)); unsafeAtomicAdd(as + 7, bf2f_hi(ud.w));
}

// ---------------------------------------------------------------------------
// SAGE layer: xout = act( x @ Wl + (agg*deg_inv) @ Wr + b ), 32 rows/block.
// 8 waves; wave nt owns a 16-col N-tile; B fragments reused across 2 M-subtiles.
// x slab staged via async-to-LDS (already bf16); agg slab converted in VALU.
// ---------------------------------------------------------------------------
__global__ __launch_bounds__(256) void sage_kernel(
    const unsigned short* __restrict__ xbf, const float* __restrict__ agg,
    const float* __restrict__ deg_inv,
    const unsigned short* __restrict__ pWl, const unsigned short* __restrict__ pWr,
    const float* __restrict__ bias, unsigned short* __restrict__ xout, int do_relu) {
  __shared__ unsigned short lx[32 * LDSTR];
  __shared__ unsigned short la[32 * LDSTR];
  int t = threadIdx.x;
  int rowBase = blockIdx.x * 32;

  // async stage: 32 rows x 256B of bf16 x, 16B chunks, 2 per thread
#pragma unroll
  for (int i = 0; i < 2; ++i) {
    int ch = t + 256 * i;
    int row = ch >> 4, seg = ch & 15;
    async_b128(xbf + (size_t)(rowBase + row) * H + seg * 8, lx + row * LDSTR + seg * 8);
  }
  // VALU stage: agg * deg_inv -> bf16 pairs (16 elems/thread)
  {
    int row = t >> 3, c0 = (t & 7) << 4;
    int grow = rowBase + row;
    float di = deg_inv[grow];
    const float* ar = agg + (size_t)grow * H + c0;
    unsigned* la32 = (unsigned*)la;
    int base = (row * LDSTR + c0) >> 1;
#pragma unroll
    for (int j = 0; j < 8; ++j)
      la32[base + j] = (unsigned)f2bf(ar[2 * j] * di) | ((unsigned)f2bf(ar[2 * j + 1] * di) << 16);
  }
  wait_async0();
  __syncthreads();

  int lane = t & 31, nt = t >> 5;
  Acc acc[2];
#pragma unroll
  for (int m = 0; m < 2; ++m)
#pragma unroll
    for (int i = 0; i < 8; ++i) acc[m].f[i] = 0.f;

#pragma unroll
  for (int kt = 0; kt < 4; ++kt) {
    v16bf bl = *((const v16bf*)(pWl + (size_t)(kt * 8 + nt) * 512) + lane);
    v16bf br = *((const v16bf*)(pWr + (size_t)(kt * 8 + nt) * 512) + lane);
#pragma unroll
    for (int mt = 0; mt < 2; ++mt) {
      v16bf ax = load_a_frag(lx + mt * 16 * LDSTR, lane, kt * 32);
      acc[mt].v = __builtin_amdgcn_wmma_f32_16x16x32_bf16(false, ax, false, bl, (short)0, acc[mt].v, false, false);
      v16bf aa = load_a_frag(la + mt * 16 * LDSTR, lane, kt * 32);
      acc[mt].v = __builtin_amdgcn_wmma_f32_16x16x32_bf16(false, aa, false, br, (short)0, acc[mt].v, false, false);
    }
  }
  int col = nt * 16 + (lane & 15), hiL = lane >> 4;
  float bb = bias[col];
#pragma unroll
  for (int mt = 0; mt < 2; ++mt)
#pragma unroll
    for (int rr = 0; rr < 8; ++rr) {
      float v = acc[mt].f[rr] + bb;
      if (do_relu) v = fmaxf(v, 0.f);
      xout[(size_t)(rowBase + mt * 16 + rr + 8 * hiL) * H + col] = f2bf(v);
    }
}

// ---------------------------------------------------------------------------
// Node head (fused): p_hat = relu(emb @ Wn1 + bn1) @ Wn2 + bn2, 32 rows/block.
// ---------------------------------------------------------------------------
__global__ __launch_bounds__(256) void node_head_kernel(
    const unsigned short* __restrict__ emb, const unsigned short* __restrict__ pWn1,
    const float* __restrict__ bn1, const float* __restrict__ Wn2,
    const float* __restrict__ bn2, float* __restrict__ p_hat) {
  __shared__ unsigned short le[32 * LDSTR];
  __shared__ float red[32];
  int t = threadIdx.x;
  int rowBase = blockIdx.x * 32;
#pragma unroll
  for (int i = 0; i < 2; ++i) {
    int ch = t + 256 * i;
    int row = ch >> 4, seg = ch & 15;
    async_b128(emb + (size_t)(rowBase + row) * H + seg * 8, le + row * LDSTR + seg * 8);
  }
  if (t < 32) red[t] = 0.f;
  wait_async0();
  __syncthreads();

  int lane = t & 31, nt = t >> 5;
  Acc acc[2];
#pragma unroll
  for (int m = 0; m < 2; ++m)
#pragma unroll
    for (int i = 0; i < 8; ++i) acc[m].f[i] = 0.f;
#pragma unroll
  for (int kt = 0; kt < 4; ++kt) {
    v16bf b = *((const v16bf*)(pWn1 + (size_t)(kt * 8 + nt) * 512) + lane);
#pragma unroll
    for (int mt = 0; mt < 2; ++mt) {
      v16bf a = load_a_frag(le + mt * 16 * LDSTR, lane, kt * 32);
      acc[mt].v = __builtin_amdgcn_wmma_f32_16x16x32_bf16(false, a, false, b, (short)0, acc[mt].v, false, false);
    }
  }
  int col = nt * 16 + (lane & 15), hiL = lane >> 4;
  float bb = bn1[col], w2 = Wn2[col];
#pragma unroll
  for (int mt = 0; mt < 2; ++mt)
#pragma unroll
    for (int rr = 0; rr < 8; ++rr) {
      float v = fmaxf(acc[mt].f[rr] + bb, 0.f) * w2;
      unsafeAtomicAdd(&red[mt * 16 + rr + 8 * hiL], v);   // ds_add_f32 across 128 cols
    }
  __syncthreads();
  if (t < 32) p_hat[rowBase + t] = red[t] + bn2[0];
}

// ---------------------------------------------------------------------------
// Edge head (fused): q_hat = relu([emb[src]|emb[dst]|feat6] @ We1 + be1) @ We2 + be2
// 64 edges/block; gathered rows staged via per-lane async-to-LDS; K=256 via
// 8 WMMA K-tiles with B reused across 4 M-subtiles (32 WMMAs/wave).
// ---------------------------------------------------------------------------
__global__ __launch_bounds__(256) void edge_head_kernel(
    const unsigned short* __restrict__ emb, const int* __restrict__ src, const int* __restrict__ dst,
    const float* __restrict__ edge_static, const float* __restrict__ q_obs,
    const int* __restrict__ q_mask,
    const unsigned short* __restrict__ pWe1, const float* __restrict__ We1,
    const float* __restrict__ be1, const float* __restrict__ We2,
    const float* __restrict__ be2, float* __restrict__ q_hat) {
  __shared__ unsigned short ls[64 * LDSTR];
  __shared__ unsigned short ld[64 * LDSTR];
  __shared__ float f6[64][6];
  __shared__ float red[64];
  int t = threadIdx.x;
  int eBase = blockIdx.x * 64;

  // async gather-stage: 2 slabs x 64 rows x 16 chunks = 2048 chunks, 8/thread
#pragma unroll
  for (int i = 0; i < 8; ++i) {
    int ch = t + 256 * i;
    int slab = ch >> 10, within = ch & 1023;
    int row = within >> 4, seg = within & 15;
    int node = (slab ? dst : src)[eBase + row];
    unsigned short* lbase = slab ? ld : ls;
    async_b128(emb + (size_t)node * H + seg * 8, lbase + row * LDSTR + seg * 8);
  }
  if (t < 64) {
    int ee = eBase + t;
#pragma unroll
    for (int j = 0; j < 4; ++j) f6[t][j] = edge_static[ee * 4 + j];
    f6[t][4] = q_obs[ee];
    f6[t][5] = q_mask[ee] ? 1.f : 0.f;
    red[t] = 0.f;
  }
  wait_async0();
  __syncthreads();

  int lane = t & 31, nt = t >> 5;
  Acc acc[4];
#pragma unroll
  for (int m = 0; m < 4; ++m)
#pragma unroll
    for (int i = 0; i < 8; ++i) acc[m].f[i] = 0.f;

#pragma unroll
  for (int kt = 0; kt < 4; ++kt) {
    v16bf bS = *((const v16bf*)(pWe1 + (size_t)(kt * 8 + nt) * 512) + lane);
    v16bf bD = *((const v16bf*)(pWe1 + (size_t)((kt + 4) * 8 + nt) * 512) + lane);
#pragma unroll
    for (int mt = 0; mt < 4; ++mt) {
      v16bf aS = load_a_frag(ls + mt * 16 * LDSTR, lane, kt * 32);
      acc[mt].v = __builtin_amdgcn_wmma_f32_16x16x32_bf16(false, aS, false, bS, (short)0, acc[mt].v, false, false);
      v16bf aD = load_a_frag(ld + mt * 16 * LDSTR, lane, kt * 32);
      acc[mt].v = __builtin_amdgcn_wmma_f32_16x16x32_bf16(false, aD, false, bD, (short)0, acc[mt].v, false, false);
    }
  }
  int col = nt * 16 + (lane & 15), hiL = lane >> 4;
  float bb = be1[col], w2 = We2[col];
  float w6[6];
#pragma unroll
  for (int j = 0; j < 6; ++j) w6[j] = We1[(256 + j) * H + col];  // rows 256..261 of [262,128]
#pragma unroll
  for (int mt = 0; mt < 4; ++mt)
#pragma unroll
    for (int rr = 0; rr < 8; ++rr) {
      int m = mt * 16 + rr + 8 * hiL;
      float z = acc[mt].f[rr] + bb;
#pragma unroll
      for (int j = 0; j < 6; ++j) z += f6[m][j] * w6[j];
      unsafeAtomicAdd(&red[m], fmaxf(z, 0.f) * w2);
    }
  __syncthreads();
  if (t < 64) q_hat[eBase + t] = red[t] + be2[0];
}

// ---------------------------------------------------------------------------
extern "C" void kernel_launch(void* const* d_in, const int* in_sizes, int n_in,
                              void* d_out, int out_size, void* d_ws, size_t ws_size,
                              hipStream_t stream) {
  const int*   edge_index  = (const int*)  d_in[0];
  const float* node_static = (const float*)d_in[1];
  const float* edge_static = (const float*)d_in[2];
  const float* p_obs       = (const float*)d_in[3];
  const float* q_obs       = (const float*)d_in[4];
  const int*   p_mask      = (const int*)  d_in[5];
  const int*   q_mask      = (const int*)  d_in[6];
  const float* W_enc = (const float*)d_in[7];
  const float* b_enc = (const float*)d_in[8];
  const float* Wl    = (const float*)d_in[9];
  const float* Wr    = (const float*)d_in[10];
  const float* b_gnn = (const float*)d_in[11];
  const float* Wn1   = (const float*)d_in[12];
  const float* bn1   = (const float*)d_in[13];
  const float* Wn2   = (const float*)d_in[14];
  const float* bn2   = (const float*)d_in[15];
  const float* We1   = (const float*)d_in[16];
  const float* be1   = (const float*)d_in[17];
  const float* We2   = (const float*)d_in[18];
  const float* be2   = (const float*)d_in[19];

  const int* src = edge_index;
  const int* dst = edge_index + NE;
  float* out = (float*)d_out;             // [p_hat (N) | q_hat (E)]

  // workspace carve-up (~103 MB), 256B-aligned slots
  char* ws = (char*)d_ws;
  size_t off = 0;
  auto carve = [&](size_t bytes) { void* p = ws + off; off = (off + bytes + 255) & ~(size_t)255; return p; };
  unsigned short* xbfA = (unsigned short*)carve((size_t)NN * H * 2);
  unsigned short* xbfB = (unsigned short*)carve((size_t)NN * H * 2);
  float* agg     = (float*)carve((size_t)NN * H * 4);
  float* deg     = (float*)carve((size_t)NN * 4);
  float* deg_inv = (float*)carve((size_t)NN * 4);
  unsigned short* pWl  = (unsigned short*)carve((size_t)3 * 32 * 512 * 2);
  unsigned short* pWr  = (unsigned short*)carve((size_t)3 * 32 * 512 * 2);
  unsigned short* pWn1 = (unsigned short*)carve((size_t)32 * 512 * 2);
  unsigned short* pWe1 = (unsigned short*)carve((size_t)64 * 512 * 2);

  // 1) pack GEMM weights to bf16 WMMA operand layout
  for (int l = 0; l < 3; ++l) {
    pack_kernel<<<(32 * 512 + 255) / 256, 256, 0, stream>>>(Wl + (size_t)l * H * H, pWl + (size_t)l * 32 * 512, 4);
    pack_kernel<<<(32 * 512 + 255) / 256, 256, 0, stream>>>(Wr + (size_t)l * H * H, pWr + (size_t)l * 32 * 512, 4);
  }
  pack_kernel<<<(32 * 512 + 255) / 256, 256, 0, stream>>>(Wn1, pWn1, 4);
  pack_kernel<<<(64 * 512 + 255) / 256, 256, 0, stream>>>(We1, pWe1, 8);   // first 256 rows

  // 2) degrees
  zero_kernel<<<(NN / 4 + 255) / 256, 256, 0, stream>>>(deg, NN / 4);
  degree_kernel<<<(NE + 255) / 256, 256, 0, stream>>>(src, dst, deg);
  deginv_kernel<<<(NN + 255) / 256, 256, 0, stream>>>(deg, deg_inv);

  // 3) encoder -> bf16 activations
  encode_kernel<<<(NN * 64 + 255) / 256, 256, 0, stream>>>(node_static, p_obs, p_mask, W_enc, b_enc,
                                                           (unsigned*)xbfA);

  // 4) 3 GraphSAGE layers
  unsigned short* xc = xbfA;
  unsigned short* xn = xbfB;
  for (int l = 0; l < 3; ++l) {
    zero_kernel<<<(NN * H / 4 + 255) / 256, 256, 0, stream>>>(agg, NN * H / 4);
    aggregate_kernel<<<(NE * 16 + 255) / 256, 256, 0, stream>>>(xc, src, dst, agg);
    sage_kernel<<<NN / 32, 256, 0, stream>>>(xc, agg, deg_inv,
                                             pWl + (size_t)l * 32 * 512, pWr + (size_t)l * 32 * 512,
                                             b_gnn + (size_t)l * H, xn, (l < 2) ? 1 : 0);
    unsigned short* tmp = xc; xc = xn; xn = tmp;
  }

  // 5) heads
  node_head_kernel<<<NN / 32, 256, 0, stream>>>(xc, pWn1, bn1, Wn2, bn2, out);
  edge_head_kernel<<<NE / 64, 256, 0, stream>>>(xc, src, dst, edge_static, q_obs, q_mask,
                                                pWe1, We1, be1, We2, be2, out + NN);
}
